// KANCouplingNet_23450521436793
// MI455X (gfx1250) — matrix operand order repaired and checked by reading, not compiled, not run on hardware
//
#include <hip/hip_runtime.h>
#include <hip/hip_bf16.h>

// ---------------------------------------------------------------------------
// KAN coupling net, fused for gfx1250 (MI455X).
//   rows N = 32*64*64 = 131072, layers 12 -> 64 -> 64 -> 24
//   Each layer: one WMMA GEMM  A(N, in*9) x W(in*9, out)  where the first in*8
//   K-slots are B-spline basis values (built on the fly, f16) and the last
//   `in` K-slots are silu(x) (base path folded into the same GEMM).
// ---------------------------------------------------------------------------

typedef __attribute__((ext_vector_type(16))) _Float16 v16h;
typedef __attribute__((ext_vector_type(8)))  _Float16 v8h;
typedef __attribute__((ext_vector_type(8)))  float    v8f;

// ---- packed-weight workspace layout (halves), written by kan_prep ----------
// layer L: Wt[n][k], k-stride RS (Kpad + 8 halves of zero pad for LDS banking)
#define RS0   136           // layer0: Kpad = 128 (in*9 = 108)
#define RS12  584           // layer1/2: Kpad = 576 (in*9 = 576)
#define WS_L0 0             // 64*136  =  8704 halves
#define WS_L1 8704          // 64*584  = 37376 halves
#define WS_L2 46080         // 32*584  = 18688 halves
#define WS_TOTAL 64768

// ---- dynamic LDS layout (bytes) -------------------------------------------
#define LDS_WT   0          // max 64*584*2 = 74752 B
#define LDS_GS   74752      // grids, 64*12 f32 = 3072 B
#define LDS_IVD  77824      // 1/(g[j+d]-g[j]) tables, 64*32 f32 = 8192 B
#define LDS_HB   86016      // activations, 8 waves * 16 rows * 68 f32 = 34816 B
#define LDS_TOTAL 120832
#define HSTR 68             // activation row stride (floats), bank-decorrelated

// silu via fast hardware reciprocal (v_rcp_f32), not the IEEE div expansion.
__device__ __forceinline__ float silu_f(float x) {
  return x * __builtin_amdgcn_rcpf(1.0f + __expf(-x));
}

// Cox-de Boor, degree 3, generic grid g[12]; ivd holds reciprocal denominators
// (slots 0..10: d=1, 11..20: d=2, 21..29: d=3). Produces 8 basis values.
// Fully branchless: step functions via cndmask, B0[j] = s[j]-s[j+1].
__device__ __forceinline__ void bspline8(float x, const float* __restrict__ g,
                                         const float* __restrict__ ivd,
                                         float* __restrict__ out) {
  float d[12];
#pragma unroll
  for (int j = 0; j < 12; ++j) d[j] = x - g[j];

  float s[12];
#pragma unroll
  for (int j = 0; j < 12; ++j) s[j] = (d[j] >= 0.0f) ? 1.0f : 0.0f;

  float B[11];
#pragma unroll
  for (int j = 0; j < 11; ++j) B[j] = s[j] - s[j + 1];   // degree 0

#pragma unroll
  for (int j = 0; j < 10; ++j) {
    float u0 = d[j]     * ivd[j];
    float u1 = d[j + 1] * ivd[j + 1];
    B[j] = u0 * B[j] + (1.0f - u1) * B[j + 1];
  }
#pragma unroll
  for (int j = 0; j < 9; ++j) {
    float u0 = d[j]     * ivd[11 + j];
    float u1 = d[j + 1] * ivd[12 + j];
    B[j] = u0 * B[j] + (1.0f - u1) * B[j + 1];
  }
#pragma unroll
  for (int j = 0; j < 8; ++j) {
    float u0 = d[j]     * ivd[21 + j];
    float u1 = d[j + 1] * ivd[22 + j];
    out[j] = u0 * B[j] + (1.0f - u1) * B[j + 1];
  }
}

// ---------------------------------------------------------------------------
// Prep kernel: pack (coef*ss | sb | 0-pad) into f16, transposed [n][k].
// ---------------------------------------------------------------------------
__global__ void kan_prep(const float* __restrict__ c0, const float* __restrict__ sb0, const float* __restrict__ ss0,
                         const float* __restrict__ c1, const float* __restrict__ sb1, const float* __restrict__ ss1,
                         const float* __restrict__ c2, const float* __restrict__ sb2, const float* __restrict__ ss2,
                         _Float16* __restrict__ wt) {
  int idx = blockIdx.x * blockDim.x + threadIdx.x;
  if (idx >= WS_TOTAL) return;
  const float *coef, *sb, *ss;
  int in, out, rs, base;
  if (idx < WS_L1)      { coef = c0; sb = sb0; ss = ss0; in = 12; out = 64; rs = RS0;  base = WS_L0; }
  else if (idx < WS_L2) { coef = c1; sb = sb1; ss = ss1; in = 64; out = 64; rs = RS12; base = WS_L1; }
  else                  { coef = c2; sb = sb2; ss = ss2; in = 64; out = 24; rs = RS12; base = WS_L2; }
  int rel = idx - base;
  int n = rel / rs, k = rel - n * rs;
  float v = 0.0f;
  if (n < out) {
    if (k < in * 8) {
      int i = k >> 3, g = k & 7;
      v = coef[(i * out + n) * 8 + g] * ss[i * out + n];
    } else if (k < in * 9) {
      int i = k - in * 8;
      v = sb[i * out + n];
    }
  }
  wt[idx] = (_Float16)v;
}

// ---------------------------------------------------------------------------
// Cooperative staging (256 threads): async global->LDS DMA (ASYNCcnt path),
// 16B per op. LDS byte address = low 32 bits of the generic shared pointer.
// ---------------------------------------------------------------------------
__device__ __forceinline__ void stage_wt_async(int tid, const _Float16* __restrict__ src,
                                               _Float16* __restrict__ dst, int n16) {
  unsigned int lds_base = (unsigned int)(unsigned long long)(void*)dst;
  unsigned long long gbase = (unsigned long long)(const void*)src;
  for (int i = tid; i < n16; i += 256) {
    unsigned int lds = lds_base + (unsigned int)i * 16u;
    unsigned long long ga = gbase + (unsigned long long)i * 16ull;
    asm volatile("global_load_async_to_lds_b128 %0, %1, off"
                 :: "v"(lds), "v"(ga) : "memory");
  }
}

__device__ __forceinline__ void async_fence() {
  asm volatile("s_wait_asynccnt 0x0" ::: "memory");
}

__device__ __forceinline__ void stage_grid(int tid, const float* __restrict__ g,
                                           float* __restrict__ gS, float* __restrict__ ivd, int in) {
  for (int i = tid; i < in * 12; i += 256) gS[i] = g[i];
  for (int idx = tid; idx < in * 30; idx += 256) {
    int i = idx / 30, s = idx - i * 30;
    const float* gr = g + i * 12;
    float d;
    if (s < 11)      d = gr[s + 1] - gr[s];
    else if (s < 21) { int j = s - 11; d = gr[j + 2] - gr[j]; }
    else             { int j = s - 21; d = gr[j + 3] - gr[j]; }
    ivd[i * 32 + s] = 1.0f / d;          // cold, keep exact IEEE divide
  }
}

// ---------------------------------------------------------------------------
// One layer: wave-level GEMM, A built in registers, B from LDS, f32 accum.
// IN = inputs, NT = output 16-col tiles, KT = K 32-tiles.
// All NT B-fragments are loaded before the WMMA chain so ds latency pipelines
// against WMMA issue (descending s_wait_dscnt instead of wait-0 per tile).
// ---------------------------------------------------------------------------
template <int IN, int NT, int KT>
__device__ __forceinline__ void layer_mm(const _Float16* __restrict__ Wt,
                                         const float* __restrict__ gS,
                                         const float* __restrict__ ivd,
                                         const float* __restrict__ hrow,
                                         int lane, v8f (&acc)[NT]) {
  constexpr int K9 = IN * 9;
  constexpr int RS = ((K9 + 31) / 32) * 32 + 8;   // 136 / 584
  const int r   = lane & 15;
  const int hlf = lane >> 4;
  for (int t = 0; t < KT; ++t) {
    // ---- B fragments first (independent ds_load_b128 pairs) ----
    v8h blo[NT], bhi[NT];
#pragma unroll
    for (int nt = 0; nt < NT; ++nt) {
      const _Float16* wp = Wt + (nt * 16 + r) * RS + t * 32 + hlf * 8;
      blo[nt] = *(const v8h*)wp;
      bhi[nt] = *(const v8h*)(wp + 16);
    }
    // ---- A fragment (VALU work overlaps outstanding ds loads) ----
    v16h a;
    if (t < IN / 4) {                     // pure spline k-tile
      int i0 = 4 * t + hlf;
      int i1 = i0 + 2;
      float b0[8], b1[8];
      bspline8(hrow[i0], gS + i0 * 12, ivd + i0 * 32, b0);
      bspline8(hrow[i1], gS + i1 * 12, ivd + i1 * 32, b1);
#pragma unroll
      for (int e = 0; e < 8; ++e) { a[e] = (_Float16)b0[e]; a[8 + e] = (_Float16)b1[e]; }
    } else {                              // silu base-path (+ zero pad) k-tile
#pragma unroll
      for (int e = 0; e < 16; ++e) {
        int K = 32 * t + ((e & 8) << 1) + 8 * hlf + (e & 7);
        int i = K - IN * 8;
        float v = (i >= 0 && i < IN) ? silu_f(hrow[i]) : 0.0f;
        a[e] = (_Float16)v;
      }
    }
    // ---- WMMA chain ----
#pragma unroll
    for (int nt = 0; nt < NT; ++nt) {
      v16h b = __builtin_shufflevector(blo[nt], bhi[nt], 0, 1, 2, 3, 4, 5, 6, 7,
                                       8, 9, 10, 11, 12, 13, 14, 15);
      acc[nt] = __builtin_amdgcn_wmma_f32_16x16x32_f16(
          false, a, false, b, (short)0, acc[nt], false, false);
    }
  }
}

// ---------------------------------------------------------------------------
// Fused 3-layer kernel. 256 threads = 8 wave32; each wave owns 16 rows.
// ---------------------------------------------------------------------------
__global__ __launch_bounds__(256) void kan_main(const float* __restrict__ x,
                                                const float* __restrict__ g0,
                                                const float* __restrict__ g1,
                                                const float* __restrict__ g2,
                                                const _Float16* __restrict__ wt,
                                                float* __restrict__ out) {
  extern __shared__ char smem[];
  _Float16* Wt = (_Float16*)(smem + LDS_WT);
  float* gS    = (float*)(smem + LDS_GS);
  float* ivd   = (float*)(smem + LDS_IVD);
  float* hbuf  = (float*)(smem + LDS_HB);

  const int tid  = threadIdx.x;
  const int w    = tid >> 5;
  const int lane = tid & 31;
  const int rowbase = blockIdx.x * 128 + w * 16;
  float* hb = hbuf + (w * 16) * HSTR;                 // this wave's 16-row strip
  const float* hrow = hb + (lane & 15) * HSTR;        // this lane's row

  // ---------------- layer 0: stage x, W0, grid0 ----------------
  stage_wt_async(tid, wt + WS_L0, Wt, 64 * RS0 * 2 / 16);
  for (int idx = lane; idx < 192; idx += 32) {        // 16 rows x 12 channels
    int r = idx / 12, c = idx - r * 12;
    int n = rowbase + r;
    hb[r * HSTR + c] = x[(((n >> 12) * 12 + c) << 12) | (n & 4095)];
  }
  stage_grid(tid, g0, gS, ivd, 12);
  async_fence();
  __syncthreads();

  // prefetch layer1 packed weights toward L2 while layer0 runs
  __builtin_prefetch((const char*)(wt + WS_L1) + tid * 256, 0, 1);

  {
    v8f acc[4];
    v8f z = {0.f, 0.f, 0.f, 0.f, 0.f, 0.f, 0.f, 0.f};
#pragma unroll
    for (int nt = 0; nt < 4; ++nt) acc[nt] = z;
    layer_mm<12, 4, 4>(Wt, gS, ivd, hrow, lane, acc);
    __syncthreads();                                  // all waves done with Wt0
#pragma unroll
    for (int nt = 0; nt < 4; ++nt)
#pragma unroll
      for (int j = 0; j < 8; ++j)
        hb[((lane >> 4) * 8 + j) * HSTR + nt * 16 + (lane & 15)] = acc[nt][j];
  }

  // ---------------- layer 1 ----------------
  stage_wt_async(tid, wt + WS_L1, Wt, 64 * RS12 * 2 / 16);
  stage_grid(tid, g1, gS, ivd, 64);
  async_fence();
  __syncthreads();
  __builtin_prefetch((const char*)(wt + WS_L2) + tid * 146, 0, 1);
  {
    v8f acc[4];
    v8f z = {0.f, 0.f, 0.f, 0.f, 0.f, 0.f, 0.f, 0.f};
#pragma unroll
    for (int nt = 0; nt < 4; ++nt) acc[nt] = z;
    layer_mm<64, 4, 18>(Wt, gS, ivd, hrow, lane, acc);
    __syncthreads();                                  // all waves done with Wt1
#pragma unroll
    for (int nt = 0; nt < 4; ++nt)
#pragma unroll
      for (int j = 0; j < 8; ++j)
        hb[((lane >> 4) * 8 + j) * HSTR + nt * 16 + (lane & 15)] = acc[nt][j];
  }

  // ---------------- layer 2 ----------------
  stage_wt_async(tid, wt + WS_L2, Wt, 32 * RS12 * 2 / 16);
  stage_grid(tid, g2, gS, ivd, 64);
  async_fence();
  __syncthreads();
  {
    v8f acc[2];
    v8f z = {0.f, 0.f, 0.f, 0.f, 0.f, 0.f, 0.f, 0.f};
#pragma unroll
    for (int nt = 0; nt < 2; ++nt) acc[nt] = z;
    layer_mm<64, 2, 18>(Wt, gS, ivd, hrow, lane, acc);
    // write (B, 24, H, W)
#pragma unroll
    for (int nt = 0; nt < 2; ++nt) {
      int c = nt * 16 + (lane & 15);
      if (c < 24) {
#pragma unroll
        for (int j = 0; j < 8; ++j) {
          int n = rowbase + (lane >> 4) * 8 + j;
          out[(((n >> 12) * 24 + c) << 12) | (n & 4095)] = acc[nt][j];
        }
      }
    }
  }
}

// ---------------------------------------------------------------------------
extern "C" void kernel_launch(void* const* d_in, const int* in_sizes, int n_in,
                              void* d_out, int out_size, void* d_ws, size_t ws_size,
                              hipStream_t stream) {
  (void)n_in; (void)out_size; (void)ws_size;
  const float* x   = (const float*)d_in[0];
  const float* g0  = (const float*)d_in[1];
  const float* c0  = (const float*)d_in[2];
  const float* sb0 = (const float*)d_in[3];
  const float* ss0 = (const float*)d_in[4];
  const float* g1  = (const float*)d_in[5];
  const float* c1  = (const float*)d_in[6];
  const float* sb1 = (const float*)d_in[7];
  const float* ss1 = (const float*)d_in[8];
  const float* g2  = (const float*)d_in[9];
  const float* c2  = (const float*)d_in[10];
  const float* sb2 = (const float*)d_in[11];
  const float* ss2 = (const float*)d_in[12];

  _Float16* wt = (_Float16*)d_ws;

  kan_prep<<<(WS_TOTAL + 255) / 256, 256, 0, stream>>>(c0, sb0, ss0, c1, sb1, ss1,
                                                       c2, sb2, ss2, wt);

  int rows = in_sizes[0] / 12;            // 131072
  int blocks = rows / 128;                // 1024 workgroups, 8 wave32 each
  kan_main<<<blocks, 256, LDS_TOTAL, stream>>>(x, g0, g1, g2, wt, (float*)d_out);
}
